// ScoreNet_9612136808491
// MI455X (gfx1250) — matrix operand order repaired and checked by reading, not compiled, not run on hardware
//
#include <hip/hip_runtime.h>
#include <hip/hip_bf16.h>

typedef __attribute__((ext_vector_type(16))) _Float16 v16h;
typedef __attribute__((ext_vector_type(8)))  _Float16 v8h;
typedef __attribute__((ext_vector_type(8)))  float    v8f;

#define B_  8
#define NV_ 512
#define C_  256
#define HR_ 192
#define N_  512
#define PH_ 16

union AFrag { v16h v; v8h h[2]; };

// A fragment: 16x32 f16 tile from row-major src (ld halves per row).
// ISA layout: lanes 0-15 hold row=lane, K {0-7,16-23}; lanes 16-31 row=lane-16, K {8-15,24-31}.
__device__ inline v16h load_a_frag(const _Float16* __restrict__ src, int row0, int k0, int ld, int lane) {
    int r  = lane & 15;
    int hs = lane >> 4;
    const _Float16* p = src + (size_t)(row0 + r) * ld + k0 + hs * 8;
    AFrag f;
    f.h[0] = *(const v8h*)(p);
    f.h[1] = *(const v8h*)(p + 16);
    return f.v;
}

// B fragment: 32x16 f16, element (k,n) = srcT[n*ld + k].
// ISA layout: lanes 0-15 hold col n=lane&15, K {0-7,16-23}; lanes 16-31 K {8-15,24-31}.
__device__ inline v16h load_b_frag(const _Float16* __restrict__ srcT, int col0, int k0, int ld, int lane) {
    int n  = lane & 15;
    int kb = k0 + (lane >> 4) * 8;
    const _Float16* p = srcT + (size_t)(col0 + n) * ld + kb;
    AFrag f;
    f.h[0] = *(const v8h*)(p);
    f.h[1] = *(const v8h*)(p + 16);
    return f.v;
}

__device__ inline float sigm(float v) { return 1.f / (1.f + expf(-v)); }

// x[b,n,c] = mean over the 2-pool of feats[:,1:,:]
__global__ void pool_kernel(const float* __restrict__ feats,
                            float* __restrict__ xf, _Float16* __restrict__ xh) {
    int tid = blockIdx.x * blockDim.x + threadIdx.x;        // < B*N*C = 2^20
    int c = tid & (C_ - 1);
    int n = (tid >> 8) & (N_ - 1);
    int b = tid >> 17;                                      // N*C = 2^17
    const float* base = feats + (size_t)b * (2 * NV_ + 1) * C_;
    float v = 0.5f * (base[(1 + 2 * n) * C_ + c] + base[(2 + 2 * n) * C_ + c]);
    xf[tid] = v;
    xh[tid] = (_Float16)v;
}

// WqT[n,k] = Wq[k,n] as f16 (and same for Wk)
__global__ void wconv_kernel(const float* __restrict__ Wq, const float* __restrict__ Wk,
                             _Float16* __restrict__ WqT, _Float16* __restrict__ WkT) {
    int tid = blockIdx.x * blockDim.x + threadIdx.x;        // < 2*HR_*C_
    int half = HR_ * C_;
    int sel = tid >= half;
    int r = sel ? tid - half : tid;
    int n = r >> 8;            // / 256
    int k = r & (C_ - 1);
    const float* W = sel ? Wk : Wq;
    _Float16* WT = sel ? WkT : WqT;
    WT[(size_t)n * C_ + k] = (_Float16)W[(size_t)k * HR_ + n];
}

// per-node projections: a_i, b_j (relu, 16-dim) and diag d
__global__ void node_kernel(const float* __restrict__ xf,
                            const float* __restrict__ Wpi, const float* __restrict__ bpi,
                            const float* __restrict__ Wpj, const float* __restrict__ bpj,
                            const float* __restrict__ dW,  const float* __restrict__ db,
                            float* __restrict__ ai, float* __restrict__ bj, float* __restrict__ dv) {
    __shared__ float xs[C_];
    int node = blockIdx.x;            // < B*N
    int t = threadIdx.x;              // 64 threads
    const float* row = xf + (size_t)node * C_;
    for (int i = t; i < C_; i += 64) xs[i] = row[i];
    __syncthreads();
    if (t < 16) {
        float acc = bpi[t];
        for (int c = 0; c < C_; ++c) acc = fmaf(xs[c], Wpi[c * PH_ + t], acc);
        ai[(size_t)node * PH_ + t] = fmaxf(acc, 0.f);
    } else if (t < 32) {
        int j = t - 16;
        float acc = bpj[j];
        for (int c = 0; c < C_; ++c) acc = fmaf(xs[c], Wpj[c * PH_ + j], acc);
        bj[(size_t)node * PH_ + j] = fmaxf(acc, 0.f);
    } else if (t == 32) {
        float acc = db[0];
        for (int c = 0; c < C_; ++c) acc = fmaf(xs[c], dW[c], acc);
        dv[node] = acc;
    }
}

// Q/K = x_f16 @ W (M=4096, K=256, N=192), one wave per 16x16 tile
__global__ void __launch_bounds__(128) proj_kernel(const _Float16* __restrict__ xh,
                                                   const _Float16* __restrict__ WqT,
                                                   const _Float16* __restrict__ WkT,
                                                   _Float16* __restrict__ Qh,
                                                   _Float16* __restrict__ Kh) {
    int wave = threadIdx.x >> 5;
    int lane = threadIdx.x & 31;
    int w = blockIdx.x * 4 + wave;     // 0..3071
    int tn = w % 12;
    int tm = w / 12;                   // 0..255
    const _Float16* WT = blockIdx.y ? WkT : WqT;
    _Float16* Out = blockIdx.y ? Kh : Qh;
    v8f c = {};
    for (int kk = 0; kk < C_; kk += 32) {
        v16h a  = load_a_frag(xh, tm * 16, kk, C_, lane);
        v16h bf = load_b_frag(WT, tn * 16, kk, C_, lane);
        c = __builtin_amdgcn_wmma_f32_16x16x32_f16(false, a, false, bf, (short)0, c, false, false);
    }
    int col   = tn * 16 + (lane & 15);
    int rbase = tm * 16 + (lane >> 4) * 8;
    #pragma unroll
    for (int v = 0; v < 8; ++v)
        Out[(size_t)(rbase + v) * HR_ + col] = (_Float16)c[v];
}

// S[b] = Q[b] @ K[b]^T / sqrt(R), one wave per 16x16 tile
__global__ void __launch_bounds__(128) score_kernel(const _Float16* __restrict__ Qh,
                                                    const _Float16* __restrict__ Kh,
                                                    float* __restrict__ out) {
    int wave = threadIdx.x >> 5;
    int lane = threadIdx.x & 31;
    int b = blockIdx.z;
    int w = blockIdx.x * 4 + wave;     // 0..1023
    int tn = w & 31;
    int tm = w >> 5;
    const _Float16* Q = Qh + (size_t)b * N_ * HR_;
    const _Float16* K = Kh + (size_t)b * N_ * HR_;
    v8f c = {};
    for (int kk = 0; kk < HR_; kk += 32) {
        if (kk + 32 < HR_) __builtin_prefetch(Q + (size_t)(tm * 16) * HR_ + kk + 32, 0, 0);
        v16h a  = load_a_frag(Q, tm * 16, kk, HR_, lane);
        v16h bf = load_b_frag(K, tn * 16, kk, HR_, lane);
        c = __builtin_amdgcn_wmma_f32_16x16x32_f16(false, a, false, bf, (short)0, c, false, false);
    }
    const float inv = 0.17677669529663687f;   // 1/sqrt(32)
    float* o = out + (size_t)b * N_ * N_;
    int col   = tn * 16 + (lane & 15);
    int rbase = tm * 16 + (lane >> 4) * 8;
    #pragma unroll
    for (int v = 0; v < 8; ++v)
        o[(size_t)(rbase + v) * N_ + col] = c[v] * inv;
}

// Fused per-pair pass: pair term + geometry MLP + angular MLP + diag + logit scale
__global__ void __launch_bounds__(256) geom_kernel(const float* __restrict__ xy,
        const float* __restrict__ gW1, const float* __restrict__ gb1,
        const float* __restrict__ gW2, const float* __restrict__ gb2,
        const float* __restrict__ aW1, const float* __restrict__ ab1,
        const float* __restrict__ aW2, const float* __restrict__ ab2,
        const float* __restrict__ ai, const float* __restrict__ bj,
        const float* __restrict__ dv,
        const float* __restrict__ p_pair, const float* __restrict__ p_geom,
        const float* __restrict__ p_angle, const float* __restrict__ p_ls,
        float* __restrict__ out) {
    __shared__ float s_gW1[12 * 32];
    __shared__ float s_gb1[32], s_gW2[32];
    __shared__ float s_aW1[6 * 16], s_ab1[16], s_aW2[16];
    __shared__ float s_ai[16];
    int t   = threadIdx.x;
    int blk = blockIdx.x;                    // b*1024 + n*2 + half
    int m = ((blk & 1) << 8) + t;            // 0..511
    int n = (blk >> 1) & (N_ - 1);
    int b = blk >> 10;
    for (int i = t; i < 384; i += 256) s_gW1[i] = gW1[i];
    if (t < 32) { s_gb1[t] = gb1[t]; s_gW2[t] = gW2[t]; }
    if (t < 96) s_aW1[t] = aW1[t];
    if (t < 16) { s_ab1[t] = ab1[t]; s_aW2[t] = aW2[t];
                  s_ai[t]  = ai[((size_t)b * N_ + n) * PH_ + t]; }
    __syncthreads();

    const float* xyb = xy + (size_t)b * N_ * 2;
    float yn = xyb[n * 2 + 0], xn = xyb[n * 2 + 1];
    float ym = xyb[m * 2 + 0], xm = xyb[m * 2 + 1];
    float dy = yn - ym, dx = xn - xm;
    bool pad = (fabsf(yn) < 1e-9f && fabsf(xn) < 1e-9f) ||
               (fabsf(ym) < 1e-9f && fabsf(xm) < 1e-9f);
    float theta = atan2f(dy, dx);
    float c1 = cosf(theta), s1 = sinf(theta);
    float r = sqrtf(dy * dy + dx * dx + 1e-8f);

    const float SP  = 1.4142135623730951f / 7.0f;       // rbf spacing
    const float GAM = 1.0f / (2.0f * SP * SP + 1e-8f);
    float feat[12];
    feat[0] = dy; feat[1] = dx;
    #pragma unroll
    for (int k = 0; k < 8; ++k) {
        float dmu = r - SP * (float)k;
        feat[2 + k] = expf(-GAM * dmu * dmu);
    }
    feat[10] = c1; feat[11] = s1;

    float G = gb2[0];
    for (int j = 0; j < 32; ++j) {
        float h = s_gb1[j];
        #pragma unroll
        for (int f = 0; f < 12; ++f) h = fmaf(feat[f], s_gW1[f * 32 + j], h);
        G = fmaf(fmaxf(h, 0.f), s_gW2[j], G);
    }
    if (pad) G = 0.f;

    float c2 = c1 * c1 - s1 * s1, s2 = 2.f * s1 * c1;
    float c4 = c2 * c2 - s2 * s2, s4 = 2.f * s2 * c2;
    float af[6] = {c1, s1, c2, s2, c4, s4};
    float Hb = ab2[0];
    for (int j = 0; j < 16; ++j) {
        float h = s_ab1[j];
        #pragma unroll
        for (int f = 0; f < 6; ++f) h = fmaf(af[f], s_aW1[f * 16 + j], h);
        Hb = fmaf(fmaxf(h, 0.f), s_aW2[j], Hb);
    }
    if (pad) Hb = 0.f;

    const float* bjm = bj + ((size_t)b * N_ + m) * PH_;
    float pair = 0.f;
    #pragma unroll
    for (int k = 0; k < 16; ++k) pair = fmaf(s_ai[k], bjm[k], pair);

    float sp = 1.5f * sigm(p_pair[0]);
    float sg = 1.5f * sigm(p_geom[0]);
    float sa = 1.5f * sigm(p_angle[0]);
    float ls = 0.5f + 2.5f * sigm(p_ls[0]);

    size_t oi = ((size_t)b * N_ + n) * N_ + m;
    float s = out[oi] + sp * pair + sg * G + sa * Hb + ((n == m) ? dv[(size_t)b * N_ + n] : 0.f);
    out[oi] = s * ls;
}

extern "C" void kernel_launch(void* const* d_in, const int* in_sizes, int n_in,
                              void* d_out, int out_size, void* d_ws, size_t ws_size,
                              hipStream_t stream) {
    (void)in_sizes; (void)n_in; (void)out_size; (void)ws_size;
    const float* feats = (const float*)d_in[0];
    const float* xy    = (const float*)d_in[1];
    const float* Wq    = (const float*)d_in[2];
    const float* Wk    = (const float*)d_in[3];
    const float* Wpi   = (const float*)d_in[4];
    const float* bpi   = (const float*)d_in[5];
    const float* Wpj   = (const float*)d_in[6];
    const float* bpj   = (const float*)d_in[7];
    const float* gW1   = (const float*)d_in[8];
    const float* gb1   = (const float*)d_in[9];
    const float* gW2   = (const float*)d_in[10];
    const float* gb2   = (const float*)d_in[11];
    const float* aW1   = (const float*)d_in[12];
    const float* ab1   = (const float*)d_in[13];
    const float* aW2   = (const float*)d_in[14];
    const float* ab2   = (const float*)d_in[15];
    const float* dW    = (const float*)d_in[16];
    const float* db    = (const float*)d_in[17];
    const float* a_pair  = (const float*)d_in[18];
    const float* a_geom  = (const float*)d_in[19];
    const float* a_angle = (const float*)d_in[20];
    const float* lsc     = (const float*)d_in[21];

    char* ws = (char*)d_ws;
    size_t off = 0;
    auto alloc = [&](size_t bytes) -> void* {
        void* p = ws + off;
        off += (bytes + 255) & ~(size_t)255;
        return p;
    };
    const size_t BN = (size_t)B_ * N_;
    float*    x_f32 = (float*)   alloc(BN * C_ * sizeof(float));
    _Float16* x_h   = (_Float16*)alloc(BN * C_ * sizeof(_Float16));
    _Float16* WqT   = (_Float16*)alloc((size_t)HR_ * C_ * sizeof(_Float16));
    _Float16* WkT   = (_Float16*)alloc((size_t)HR_ * C_ * sizeof(_Float16));
    _Float16* Qh    = (_Float16*)alloc(BN * HR_ * sizeof(_Float16));
    _Float16* Kh    = (_Float16*)alloc(BN * HR_ * sizeof(_Float16));
    float*    aiw   = (float*)   alloc(BN * PH_ * sizeof(float));
    float*    bjw   = (float*)   alloc(BN * PH_ * sizeof(float));
    float*    dvw   = (float*)   alloc(BN * sizeof(float));

    float* out = (float*)d_out;

    pool_kernel <<<4096, 256, 0, stream>>>(feats, x_f32, x_h);
    wconv_kernel<<<(2 * HR_ * C_) / 256, 256, 0, stream>>>(Wq, Wk, WqT, WkT);
    node_kernel <<<(int)BN, 64, 0, stream>>>(x_f32, Wpi, bpi, Wpj, bpj, dW, db, aiw, bjw, dvw);
    proj_kernel <<<dim3(768, 2), 128, 0, stream>>>(x_h, WqT, WkT, Qh, Kh);
    score_kernel<<<dim3(256, 1, 8), 128, 0, stream>>>(Qh, Kh, out);
    geom_kernel <<<8192, 256, 0, stream>>>(xy, gW1, gb1, gW2, gb2, aW1, ab1, aW2, ab2,
                                           aiw, bjw, dvw, a_pair, a_geom, a_angle, lsc, out);
}
